// KNNRegressor_7215545057604
// MI455X (gfx1250) — matrix operand order repaired and checked by reading, not compiled, not run on hardware
//
#include <hip/hip_runtime.h>
#include <hip/hip_bf16.h>

// KNN regressor: X_train [65536,128] f32, X_test [4096,128] f32, y_train [65536] f32.
// out[q] = mean of y_train over the 5 nearest train points (Euclidean).
//
// MI455X strategy:
//  - Ranking key: ||b||^2 - 2 a.b (a^2 constant per query row, sqrt monotone).
//  - f16 WMMA (v_wmma_f32_16x16x32_f16) with f32 accumulate; X_train f16 (16.8 MB)
//    lives in the 192 MB L2, so the kernel is compute/issue bound, not HBM bound.
//  - Fused top-5 with a threshold-gated slow path; fast path is ~8 VALU ops that
//    co-execute with the XDL pipe. Insertion is branch-free (cmp+cndmask) to keep
//    register pressure flat (round-2 nested-branch version spilled to scratch).
//  - Software-pipelined B fragments (ping-pong, unroll x2): loads for chunk n+1
//    are in flight during the WMMAs of chunk n.
//  - 8-way N-split -> 2048 waves for latency hiding; the 8 waves of each block
//    share one B stream (same segment) for WGP$/L2 reuse.
//
// Workspace layout (total ~19.4 MB):
//   [0)         Xtest_f16  : 4096*128*2  = 1,048,576 B
//   [1 MB)      Xtrain_f16 : 65536*128*2 = 16,777,216 B
//   [+16 MB)    b2         : 65536*4     = 262,144 B
//   then        pval       : 4096*8*5*4  = 655,360 B
//   then        pidx       : 4096*8*5*4  = 655,360 B

typedef __attribute__((ext_vector_type(16))) _Float16 v16h;
typedef __attribute__((ext_vector_type(8)))  _Float16 v8h;
typedef __attribute__((ext_vector_type(4)))  _Float16 v4h;
typedef __attribute__((ext_vector_type(8)))  float    v8f;

#define NT     65536
#define NQ     4096
#define DIMS   128
#define NSEG   8
#define SEGLEN (NT / NSEG)      // 8192
#define NCH    (SEGLEN / 16)    // 512 chunks (even, so unroll-by-2 is exact)
#define KNN    5

union HF { v16h v; v8h h[2]; };

// Branch-free sorted insertion into t0<=t1<=t2<=t3<=t4 (5 v_cmp + cndmask chain;
// no EXEC manipulation, no long live ranges across divergent control flow).
// Strict < keeps the earlier index on ties (matches increasing-index scan order).
__device__ __forceinline__ void insert5(float s, int idx,
    float& t0, float& t1, float& t2, float& t3, float& t4,
    int& i0, int& i1, int& i2, int& i3, int& i4) {
  bool c0 = s < t0, c1 = s < t1, c2 = s < t2, c3 = s < t3, c4 = s < t4;
  float nt4 = c4 ? (c3 ? t3 : s) : t4;  int ni4 = c4 ? (c3 ? i3 : idx) : i4;
  float nt3 = c3 ? (c2 ? t2 : s) : t3;  int ni3 = c3 ? (c2 ? i2 : idx) : i3;
  float nt2 = c2 ? (c1 ? t1 : s) : t2;  int ni2 = c2 ? (c1 ? i1 : idx) : i2;
  float nt1 = c1 ? (c0 ? t0 : s) : t1;  int ni1 = c1 ? (c0 ? i0 : idx) : i1;
  float nt0 = c0 ? s : t0;              int ni0 = c0 ? idx : i0;
  t0 = nt0; t1 = nt1; t2 = nt2; t3 = nt3; t4 = nt4;
  i0 = ni0; i1 = ni1; i2 = ni2; i3 = ni3; i4 = ni4;
}

// ---- prep: X_test f32 -> f16 ------------------------------------------------
__global__ void prep_test_kernel(const float* __restrict__ x, _Float16* __restrict__ o) {
  int i = blockIdx.x * blockDim.x + threadIdx.x;      // one thread per 4 elems
  if (i < NQ * DIMS / 4) {
    float4 v = ((const float4*)x)[i];
    v4h h = { (_Float16)v.x, (_Float16)v.y, (_Float16)v.z, (_Float16)v.w };
    ((v4h*)o)[i] = h;
  }
}

// ---- prep: X_train f32 -> f16 + row squared norms (wave per row) -----------
__global__ void prep_train_kernel(const float* __restrict__ x, _Float16* __restrict__ o,
                                  float* __restrict__ b2) {
  int w    = (blockIdx.x * blockDim.x + threadIdx.x) >> 5;
  int lane = threadIdx.x & 31;
  if (w >= NT) return;
  float4 v = *(const float4*)(x + (size_t)w * DIMS + lane * 4);
  float ss = v.x * v.x + v.y * v.y + v.z * v.z + v.w * v.w;
  v4h h = { (_Float16)v.x, (_Float16)v.y, (_Float16)v.z, (_Float16)v.w };
  *(v4h*)(o + (size_t)w * DIMS + lane * 4) = h;
#pragma unroll
  for (int off = 16; off > 0; off >>= 1) ss += __shfl_xor(ss, off, 32);
  if (lane == 0) b2[w] = ss;
}

// ---- main: fused WMMA GEMM + per-query top-5 over one N-segment ------------
__global__ void __launch_bounds__(256)
knn_topk_kernel(const _Float16* __restrict__ xte16,
                const _Float16* __restrict__ xtr16,
                const float* __restrict__ b2,
                float* __restrict__ pval, int* __restrict__ pidx) {
  __shared__ float lds[8 * 272];                 // per wave: 16x16 scores + 16 thr
  const int lane  = threadIdx.x & 31;
  const int wid   = threadIdx.x >> 5;
  const int gw    = blockIdx.x * 8 + wid;        // 0..2047
  const int qtile = gw & 255;
  const int seg   = gw >> 8;                     // 0..7; uniform within a block
  const int qbase = qtile * 16;
  const int col   = lane & 15;
  const int hi    = lane >> 4;
  float* wl = lds + wid * 272;

  // A fragments: 16 queries x 128 dims, ISA 16-bit A 16x32 layout, K-chunks of 32.
  const _Float16* ap = xte16 + (size_t)(qbase + col) * DIMS;
  v16h a[4];
#pragma unroll
  for (int kc = 0; kc < 4; ++kc) {
    HF f;
    f.h[0] = *(const v8h*)(ap + kc * 32 + hi * 8);
    f.h[1] = *(const v8h*)(ap + kc * 32 + 16 + hi * 8);
    a[kc] = f.v;
  }

  const float INF = __builtin_inff();
  float t0, t1, t2, t3, t4;
  int   i0, i1, i2, i3, i4;
  t0 = t1 = t2 = t3 = t4 = INF;
  i0 = i1 = i2 = i3 = i4 = 0;
  float thrLane = INF;                            // max kth-score over this lane's 8 rows

  int nb = seg * SEGLEN;
  // B layout: lane holds column n = nb+col, K = hi*8 + 0..7 and 16 + hi*8 + 0..7.
  const _Float16* bp  = xtr16 + (size_t)(nb + col) * DIMS + hi * 8;
  const float*    b2p = b2 + nb + col;

  // top-5 maintenance on one 16x16 score tile held in accumulator c
  auto topk_step = [&](const v8f& c, float b2v, int nbase) {
    float cmax = c[0];
#pragma unroll
    for (int r = 1; r < 8; ++r) cmax = fmaxf(cmax, c[r]);
    float smin = fmaf(-2.f, cmax, b2v);           // min score among lane's 8 rows
    if (__any((int)(smin < thrLane))) {
#pragma unroll
      for (int r = 0; r < 8; ++r)
        wl[(r + 8 * hi) * 16 + col] = fmaf(-2.f, c[r], b2v);
      __builtin_amdgcn_wave_barrier();
      asm volatile("s_wait_dscnt 0" ::: "memory");   // LDS in-order per wave
      if (lane < 16) {
        for (int j = 0; j < 16; ++j)
          insert5(wl[lane * 16 + j], nbase + j, t0, t1, t2, t3, t4, i0, i1, i2, i3, i4);
        wl[256 + lane] = t4;                          // row's kth-smallest so far
      }
      __builtin_amdgcn_wave_barrier();
      asm volatile("s_wait_dscnt 0" ::: "memory");
      float m = wl[256 + hi * 8];
#pragma unroll
      for (int k = 1; k < 8; ++k) m = fmaxf(m, wl[256 + hi * 8 + k]);
      thrLane = m;
    }
  };

#define LOADB(d, base, off)                                        \
  do {                                                             \
    (d).h[0] = *(const v8h*)((base) + (off));                      \
    (d).h[1] = *(const v8h*)((base) + (off) + 16);                 \
  } while (0)

  // ping-pong B fragments: f holds chunk n, g holds chunk n+1 (in flight)
  HF f0, f1, f2, f3, g0, g1, g2, g3;
  float b2f, b2g;
  LOADB(f0, bp, 0); LOADB(f1, bp, 32); LOADB(f2, bp, 64); LOADB(f3, bp, 96);
  b2f = *b2p;

  for (int chunk = 0; chunk < NCH; chunk += 2) {
    // even half: prefetch chunk+1 into g, compute chunk from f
    const _Float16* bp1 = bp + 16 * DIMS;
    LOADB(g0, bp1, 0); LOADB(g1, bp1, 32); LOADB(g2, bp1, 64); LOADB(g3, bp1, 96);
    b2g = b2p[16];
    {
      v8f c = {0.f, 0.f, 0.f, 0.f, 0.f, 0.f, 0.f, 0.f};
      c = __builtin_amdgcn_wmma_f32_16x16x32_f16(false, a[0], false, f0.v, (short)0, c, false, false);
      c = __builtin_amdgcn_wmma_f32_16x16x32_f16(false, a[1], false, f1.v, (short)0, c, false, false);
      c = __builtin_amdgcn_wmma_f32_16x16x32_f16(false, a[2], false, f2.v, (short)0, c, false, false);
      c = __builtin_amdgcn_wmma_f32_16x16x32_f16(false, a[3], false, f3.v, (short)0, c, false, false);
      topk_step(c, b2f, nb);
    }
    // odd half: prefetch chunk+2 into f, compute chunk+1 from g
    // (last prefetch reads 4 KB past the segment: still inside the workspace)
    const _Float16* bp2 = bp + 32 * DIMS;
    LOADB(f0, bp2, 0); LOADB(f1, bp2, 32); LOADB(f2, bp2, 64); LOADB(f3, bp2, 96);
    b2f = b2p[32];
    {
      v8f c = {0.f, 0.f, 0.f, 0.f, 0.f, 0.f, 0.f, 0.f};
      c = __builtin_amdgcn_wmma_f32_16x16x32_f16(false, a[0], false, g0.v, (short)0, c, false, false);
      c = __builtin_amdgcn_wmma_f32_16x16x32_f16(false, a[1], false, g1.v, (short)0, c, false, false);
      c = __builtin_amdgcn_wmma_f32_16x16x32_f16(false, a[2], false, g2.v, (short)0, c, false, false);
      c = __builtin_amdgcn_wmma_f32_16x16x32_f16(false, a[3], false, g3.v, (short)0, c, false, false);
      topk_step(c, b2g, nb + 16);
    }
    nb  += 32;
    bp  += 32 * DIMS;
    b2p += 32;
  }
#undef LOADB

  if (lane < 16) {
    int q = qbase + lane;
    int base = (q * NSEG + seg) * KNN;
    pval[base + 0] = t0; pval[base + 1] = t1; pval[base + 2] = t2;
    pval[base + 3] = t3; pval[base + 4] = t4;
    pidx[base + 0] = i0; pidx[base + 1] = i1; pidx[base + 2] = i2;
    pidx[base + 3] = i3; pidx[base + 4] = i4;
  }
}

// ---- merge: 8 segment top-5s -> global top-5 -> label mean -----------------
__global__ void merge_kernel(const float* __restrict__ pval, const int* __restrict__ pidx,
                             const float* __restrict__ y, float* __restrict__ out) {
  int q = blockIdx.x * blockDim.x + threadIdx.x;
  if (q >= NQ) return;
  const float INF = __builtin_inff();
  float t0, t1, t2, t3, t4;
  int   i0, i1, i2, i3, i4;
  t0 = t1 = t2 = t3 = t4 = INF;
  i0 = i1 = i2 = i3 = i4 = 0;
  int base = q * NSEG * KNN;
  for (int j = 0; j < NSEG * KNN; ++j)
    insert5(pval[base + j], pidx[base + j], t0, t1, t2, t3, t4, i0, i1, i2, i3, i4);
  out[q] = (y[i0] + y[i1] + y[i2] + y[i3] + y[i4]) * 0.2f;
}

extern "C" void kernel_launch(void* const* d_in, const int* in_sizes, int n_in,
                              void* d_out, int out_size, void* d_ws, size_t ws_size,
                              hipStream_t stream) {
  (void)in_sizes; (void)n_in; (void)out_size; (void)ws_size;
  const float* X_train = (const float*)d_in[0];
  const float* X_test  = (const float*)d_in[1];
  const float* y_train = (const float*)d_in[2];
  float* out = (float*)d_out;

  char* ws = (char*)d_ws;
  _Float16* xte16 = (_Float16*)ws;                                 // 1,048,576 B
  _Float16* xtr16 = (_Float16*)(ws + (size_t)1048576);             // 16,777,216 B
  float*    b2    = (float*)(ws + (size_t)1048576 + 16777216);     // 262,144 B
  float*    pval  = (float*)(ws + (size_t)1048576 + 16777216 + 262144);
  int*      pidx  = (int*)  (ws + (size_t)1048576 + 16777216 + 262144 + 655360);

  // 1) X_test -> f16
  prep_test_kernel<<<(NQ * DIMS / 4 + 255) / 256, 256, 0, stream>>>(X_test, xte16);
  // 2) X_train -> f16 + row norms (wave per row: 65536 waves)
  prep_train_kernel<<<NT / 8, 256, 0, stream>>>(X_train, xtr16, b2);
  // 3) fused WMMA + top-5: 2048 waves = 256 blocks of 8 waves
  knn_topk_kernel<<<256, 256, 0, stream>>>(xte16, xtr16, b2, pval, pidx);
  // 4) merge partials + average labels
  merge_kernel<<<(NQ + 255) / 256, 256, 0, stream>>>(pval, pidx, y_train, out);
}
